// ViG_61168924230435
// MI455X (gfx1250) — matrix operand order repaired and checked by reading, not compiled, not run on hardware
//
#include <hip/hip_runtime.h>

// ---------------------------------------------------------------------------
// ViG forward for MI455X (gfx1250, wave32). All conv1x1 GEMMs and the kNN
// Gram matrix run on v_wmma_f32_16x16x32_f16 (f16 in, f32 accumulate).
// Activations live in transposed (B, N, K) f16 form so every WMMA operand
// fragment is a contiguous 128-bit load; GEMM epilogues emit the next layer's
// transposed-f16 input directly. Each wave computes a 32x32 output tile
// (2 A-frags x 2 B-frags -> 4 WMMAs, 2 b128 loads per WMMA).
// ---------------------------------------------------------------------------

namespace {
constexpr int Bb   = 4;
constexpr int Dd   = 192;
constexpr int Nn   = 2500;   // 50*50
constexpr int NBLK = 12;
constexpr int KMAX = 18;
constexpr float BN_EPS = 1e-5f;
}

typedef __attribute__((ext_vector_type(16))) _Float16 v16h;
typedef __attribute__((ext_vector_type(8)))  _Float16 v8h;
typedef __attribute__((ext_vector_type(8)))  float    v8f;
typedef __attribute__((ext_vector_type(4)))  _Float16 v4h;

__device__ __forceinline__ float gelu_exact(float v) {
  return 0.5f * v * (1.0f + erff(v * 0.7071067811865476f));
}

__device__ __forceinline__ v16h cat8(v8h a, v8h b) {
  return __builtin_shufflevector(a, b, 0, 1, 2, 3, 4, 5, 6, 7,
                                 8, 9, 10, 11, 12, 13, 14, 15);
}

// ------------------------------------------------ f32 -> f16 (x4 vectorized)
__global__ void k_cvt16v(const float* __restrict__ s, _Float16* __restrict__ d, int n4) {
  int i = blockIdx.x * 256 + threadIdx.x;
  if (i >= n4) return;
  const float4 v = ((const float4*)s)[i];
  v4h o; o[0] = (_Float16)v.x; o[1] = (_Float16)v.y; o[2] = (_Float16)v.z; o[3] = (_Float16)v.w;
  ((v4h*)d)[i] = o;
}

// ------------------------------ (B, C, N) f32 -> (B, N, C) f16 (input only)
__global__ void k_cvtT(const float* __restrict__ X, _Float16* __restrict__ Xt, int C) {
  int t = blockIdx.x * 256 + threadIdx.x;
  if (t >= Bb * Nn) return;
  const int b = t / Nn, n = t % Nn;
  const float* __restrict__ xp = X + (size_t)b * C * Nn + n;
  _Float16* __restrict__ op = Xt + (size_t)t * C;
  for (int c = 0; c < C; ++c) op[c] = (_Float16)xp[(size_t)c * Nn];
}

// ---------------------------------------------------------------------------
// One wave computes a 32(M) x 32(N) tile with 4 accumulators.
//   W  : (M, K) f16 row-major
//   Xt : (B, Nn, K) f16 row-major (transposed activations)
//   out : optional f32 (B, M, Nn); outT : optional f16 (B, Nn, M)
// v = act((acc + bias[o]) * s[o] + t[o]) + res
// ---------------------------------------------------------------------------
__global__ __launch_bounds__(32)
void k_gemm_bn(const _Float16* __restrict__ W, const _Float16* __restrict__ Xt,
               const float* __restrict__ bias, const float* __restrict__ bnp,
               const float* __restrict__ res, float* __restrict__ out,
               _Float16* __restrict__ outT, int M, int K, int act)
{
  const int lane = threadIdx.x;
  const int h16  = lane >> 4;          // 0/1
  const int l15  = lane & 15;
  const int n0   = blockIdx.x * 32;
  const int m0   = blockIdx.y * 32;
  const int b    = blockIdx.z;
  const int nA   = n0 + l15;
  const int nB   = n0 + 16 + l15;
  const int nAc  = (nA < Nn) ? nA : (Nn - 1);
  const int nBc  = (nB < Nn) ? nB : (Nn - 1);
  const int ko   = h16 * 8;            // A-frag K sub-offset
  const int kr   = h16 * 16;           // unused; B-frags use ko too

  const _Float16* __restrict__ Ar0 = W + (size_t)(m0 + l15) * K;
  const _Float16* __restrict__ Ar1 = W + (size_t)(m0 + 16 + l15) * K;
  const _Float16* __restrict__ XA  = Xt + ((size_t)b * Nn + nAc) * K;
  const _Float16* __restrict__ XB  = Xt + ((size_t)b * Nn + nBc) * K;

  v8f acc00 = {}, acc01 = {}, acc10 = {}, acc11 = {};
  for (int kb = 0; kb < K; kb += 32) {
    const v16h a0 = cat8(*(const v8h*)(Ar0 + kb + ko),
                         *(const v8h*)(Ar0 + kb + ko + 16));
    const v16h a1 = cat8(*(const v8h*)(Ar1 + kb + ko),
                         *(const v8h*)(Ar1 + kb + ko + 16));
    const v16h b0 = cat8(*(const v8h*)(XA + kb + ko),
                         *(const v8h*)(XA + kb + ko + 16));
    const v16h b1 = cat8(*(const v8h*)(XB + kb + ko),
                         *(const v8h*)(XB + kb + ko + 16));
    acc00 = __builtin_amdgcn_wmma_f32_16x16x32_f16(false, a0, false, b0,
                                                   (short)0, acc00, false, false);
    acc01 = __builtin_amdgcn_wmma_f32_16x16x32_f16(false, a0, false, b1,
                                                   (short)0, acc01, false, false);
    acc10 = __builtin_amdgcn_wmma_f32_16x16x32_f16(false, a1, false, b0,
                                                   (short)0, acc10, false, false);
    acc11 = __builtin_amdgcn_wmma_f32_16x16x32_f16(false, a1, false, b1,
                                                   (short)0, acc11, false, false);
  }

  // -------- epilogue: BN scale/shift per output channel, act, residual
  float sc[2][8], sh[2][8];
#pragma unroll
  for (int rm = 0; rm < 2; ++rm) {
#pragma unroll
    for (int r = 0; r < 8; ++r) {
      const int o = m0 + rm * 16 + h16 * 8 + r;
      const float ga = bnp[o], be = bnp[M + o], mu = bnp[2 * M + o], va = bnp[3 * M + o];
      const float s = ga * rsqrtf(va + BN_EPS);
      sc[rm][r] = s;
      sh[rm][r] = be - mu * s + bias[o] * s;
    }
  }
#pragma unroll 1
  for (int cn = 0; cn < 2; ++cn) {
    const int n = (cn == 0) ? nA : nB;
    if (n >= Nn) continue;
#pragma unroll
    for (int rm = 0; rm < 2; ++rm) {
      const v8f acc = (cn == 0) ? (rm == 0 ? acc00 : acc10)
                                : (rm == 0 ? acc01 : acc11);
      float vv[8];
#pragma unroll
      for (int r = 0; r < 8; ++r) {
        float v = acc[r] * sc[rm][r] + sh[rm][r];
        if (act) v = gelu_exact(v);
        if (res) v += res[((size_t)b * M + (m0 + rm * 16 + h16 * 8 + r)) * Nn + n];
        vv[r] = v;
      }
      if (out) {
#pragma unroll
        for (int r = 0; r < 8; ++r)
          out[((size_t)b * M + (m0 + rm * 16 + h16 * 8 + r)) * Nn + n] = vv[r];
      }
      if (outT) {
        v8h p;
#pragma unroll
        for (int r = 0; r < 8; ++r) p[r] = (_Float16)vv[r];
        *(v8h*)(outT + ((size_t)b * Nn + n) * M + m0 + rm * 16 + h16 * 8) = p;
      }
    }
  }
}

// ------------------------------------------------- channel-normalize + (N,C)
__global__ void k_normalize(const float* __restrict__ G, _Float16* __restrict__ Gt,
                            float* __restrict__ sq)
{
  int t = blockIdx.x * 128 + threadIdx.x;
  if (t >= Bb * Nn) return;
  const int b = t / Nn, n = t - b * Nn;
  const float* __restrict__ gp = G + (size_t)b * Dd * Nn + n;
  float s = 0.f;
  for (int c = 0; c < Dd; ++c) { float v = gp[(size_t)c * Nn]; s += v * v; }
  const float inv = 1.0f / fmaxf(sqrtf(s), 1e-12f);
  _Float16* __restrict__ op = Gt + (size_t)t * Dd;
  float s2 = 0.f;
  for (int c = 0; c < Dd; ++c) {
    float v = gp[(size_t)c * Nn] * inv;
    s2 += v * v;
    op[c] = (_Float16)v;
  }
  sq[t] = s2;
}

// ---------------------------------------- dist via WMMA Gram (32 x 32 tiles)
// dist[n][m] = sq[n] - 2*<xn_n, xn_m> + sq[m];  Gt is (N, C) row-major f16
__global__ __launch_bounds__(32)
void k_dist(const _Float16* __restrict__ Gt, const float* __restrict__ sq,
            float* __restrict__ dist)
{
  const int lane = threadIdx.x;
  const int h16  = lane >> 4;
  const int l15  = lane & 15;
  const int r0   = blockIdx.y * 32;   // rows n (two 16-groups)
  const int c0   = blockIdx.x * 32;   // cols m (two 16-groups)
  const int ko   = h16 * 8;
  const int kr   = h16 * 16;
  int ar0 = r0 + l15;      if (ar0 >= Nn) ar0 = Nn - 1;
  int ar1 = r0 + 16 + l15; if (ar1 >= Nn) ar1 = Nn - 1;
  int mA  = c0 + l15;      if (mA  >= Nn) mA  = Nn - 1;
  int mB  = c0 + 16 + l15; if (mB  >= Nn) mB  = Nn - 1;
  const _Float16* __restrict__ Ap0 = Gt + (size_t)ar0 * Dd;
  const _Float16* __restrict__ Ap1 = Gt + (size_t)ar1 * Dd;
  const _Float16* __restrict__ BpA = Gt + (size_t)mA * Dd;
  const _Float16* __restrict__ BpB = Gt + (size_t)mB * Dd;
  v8f acc00 = {}, acc01 = {}, acc10 = {}, acc11 = {};
  for (int kb = 0; kb < Dd; kb += 32) {
    const v16h a0 = cat8(*(const v8h*)(Ap0 + kb + ko),
                         *(const v8h*)(Ap0 + kb + ko + 16));
    const v16h a1 = cat8(*(const v8h*)(Ap1 + kb + ko),
                         *(const v8h*)(Ap1 + kb + ko + 16));
    const v16h b0 = cat8(*(const v8h*)(BpA + kb + ko),
                         *(const v8h*)(BpA + kb + ko + 16));
    const v16h b1 = cat8(*(const v8h*)(BpB + kb + ko),
                         *(const v8h*)(BpB + kb + ko + 16));
    acc00 = __builtin_amdgcn_wmma_f32_16x16x32_f16(false, a0, false, b0,
                                                   (short)0, acc00, false, false);
    acc01 = __builtin_amdgcn_wmma_f32_16x16x32_f16(false, a0, false, b1,
                                                   (short)0, acc01, false, false);
    acc10 = __builtin_amdgcn_wmma_f32_16x16x32_f16(false, a1, false, b0,
                                                   (short)0, acc10, false, false);
    acc11 = __builtin_amdgcn_wmma_f32_16x16x32_f16(false, a1, false, b1,
                                                   (short)0, acc11, false, false);
  }
#pragma unroll 1
  for (int cn = 0; cn < 2; ++cn) {
    const int col = c0 + cn * 16 + l15;
    if (col >= Nn) continue;
    const float sqc = sq[col];
#pragma unroll
    for (int rm = 0; rm < 2; ++rm) {
      const v8f acc = (cn == 0) ? (rm == 0 ? acc00 : acc10)
                                : (rm == 0 ? acc01 : acc11);
#pragma unroll
      for (int r = 0; r < 8; ++r) {
        const int row = r0 + rm * 16 + h16 * 8 + r;
        if (row < Nn)
          dist[(size_t)row * Nn + col] = sq[row] - 2.0f * acc[r] + sqc;
      }
    }
  }
}

// ------------------------------------------------- top-(k*dil) + dilation ::dil
__global__ void k_topk(const float* __restrict__ dist, int* __restrict__ idx,
                       int kk, int dil)
{
  int n = blockIdx.x * 128 + threadIdx.x;
  if (n >= Nn) return;
  const int kp = kk * dil;          // <= 54
  float bd[56];
  int   bi[56];
  for (int j = 0; j < kp; ++j) { bd[j] = 3.4e38f; bi[j] = 0; }
  const float* __restrict__ dr = dist + (size_t)n * Nn;
  for (int m = 0; m < Nn; ++m) {
    float d = dr[m];
    if (d < bd[kp - 1]) {
      int p = kp - 1;
      while (p > 0 && bd[p - 1] > d) { bd[p] = bd[p - 1]; bi[p] = bi[p - 1]; --p; }
      bd[p] = d; bi[p] = m;
    }
  }
  int* __restrict__ op = idx + (size_t)n * KMAX;
  for (int j = 0; j < kk; ++j) op[j] = bi[j * dil];
}

// ------------- MRConv aggregate, writes transposed f16 (B, N, 2C) directly
// Tt[b][n][2c] = g;  Tt[b][n][2c+1] = max_j (g[idx_j] - g)
__global__ void k_mrconv(const float* __restrict__ G, const int* __restrict__ idx,
                         _Float16* __restrict__ Tt, int kk)
{
  int t = blockIdx.x * 256 + threadIdx.x;
  if (t >= Bb * Dd * Nn) return;
  const int n = t % Nn;
  const int c = (t / Nn) % Dd;
  const int b = t / (Nn * Dd);
  const float* __restrict__ gp = G + ((size_t)b * Dd + c) * Nn;
  const int* __restrict__ ip = idx + ((size_t)b * Nn + n) * KMAX;
  const float xv = gp[n];
  float mx = -3.4e38f;
  for (int j = 0; j < kk; ++j) mx = fmaxf(mx, gp[ip[j]] - xv);
  _Float16* __restrict__ tp = Tt + ((size_t)b * Nn + n) * (2 * Dd) + 2 * c;
  tp[0] = (_Float16)xv;
  tp[1] = (_Float16)mx;
}

// ---------------------------------------------------------------------------
extern "C" void kernel_launch(void* const* d_in, const int* in_sizes, int n_in,
                              void* d_out, int out_size, void* d_ws, size_t ws_size,
                              hipStream_t stream)
{
  (void)in_sizes; (void)n_in; (void)out_size; (void)ws_size;
  const float* x       = (const float*)d_in[0];
  const float* fc1_w   = (const float*)d_in[1];
  const float* fc1_b   = (const float*)d_in[2];
  const float* fc1_bn  = (const float*)d_in[3];
  const float* gc_w    = (const float*)d_in[4];
  const float* gc_b    = (const float*)d_in[5];
  const float* gc_bn   = (const float*)d_in[6];
  const float* fc2_w   = (const float*)d_in[7];
  const float* fc2_b   = (const float*)d_in[8];
  const float* fc2_bn  = (const float*)d_in[9];
  const float* ffn1_w  = (const float*)d_in[10];
  const float* ffn1_b  = (const float*)d_in[11];
  const float* ffn1_bn = (const float*)d_in[12];
  const float* ffn2_w  = (const float*)d_in[13];
  const float* ffn2_b  = (const float*)d_in[14];
  const float* ffn2_bn = (const float*)d_in[15];

  // -------- carve scratch
  char* ws = (char*)d_ws;
  size_t off = 0;
  auto carve = [&](size_t bytes) -> char* {
    char* p = ws + off;
    off += (bytes + 255) & ~(size_t)255;
    return p;
  };
  const size_t FN = (size_t)Bb * Nn;            // 10000 points
  float*    hA   = (float*)carve(FN * Dd * 4);
  float*    hB   = (float*)carve(FN * Dd * 4);
  float*    G    = (float*)carve(FN * Dd * 4);
  float*    dist = (float*)carve((size_t)Nn * Nn * 4);        // 25MB
  _Float16* XtA  = (_Float16*)carve(FN * Dd * 2);             // h (transposed)
  _Float16* Gt   = (_Float16*)carve(FN * Dd * 2);             // normalized
  _Float16* Tt   = (_Float16*)carve(FN * 2 * Dd * 2);         // mrconv out
  _Float16* Yt   = (_Float16*)carve(FN * 2 * Dd * 2);         // gc out
  _Float16* HtB  = (_Float16*)carve(FN * Dd * 2);             // fc2 out
  _Float16* Ft   = (_Float16*)carve(FN * 4 * Dd * 2);         // ffn1 out
  float*    sq   = (float*)carve(FN * 4);
  int*      idxb = (int*)carve(FN * KMAX * 4);
  // full-network f16 weights (converted once)
  _Float16* Wfc1  = (_Float16*)carve((size_t)NBLK * Dd * Dd * 2);
  _Float16* Wgc   = (_Float16*)carve((size_t)NBLK * 2 * Dd * 2 * Dd * 2);
  _Float16* Wfc2  = (_Float16*)carve((size_t)NBLK * Dd * 2 * Dd * 2);
  _Float16* Wffn1 = (_Float16*)carve((size_t)NBLK * 4 * Dd * Dd * 2);
  _Float16* Wffn2 = (_Float16*)carve((size_t)NBLK * Dd * 4 * Dd * 2);

  static const int KS[NBLK]  = {9, 9, 10, 11, 12, 13, 13, 14, 15, 16, 17, 18};
  static const int DIL[NBLK] = {1, 1, 1, 1, 2, 2, 2, 2, 3, 3, 3, 3};

  const int NT32 = (Nn + 31) / 32;  // 79
  const dim3 blk32(32);

  auto cvtw = [&](const float* s, _Float16* d, size_t n) {
    int n4 = (int)(n / 4);
    k_cvt16v<<<dim3((n4 + 255) / 256), dim3(256), 0, stream>>>(s, d, n4);
  };
  // weights once, for all 12 blocks
  cvtw(fc1_w,  Wfc1,  (size_t)NBLK * Dd * Dd);
  cvtw(gc_w,   Wgc,   (size_t)NBLK * 2 * Dd * 2 * Dd);
  cvtw(fc2_w,  Wfc2,  (size_t)NBLK * Dd * 2 * Dd);
  cvtw(ffn1_w, Wffn1, (size_t)NBLK * 4 * Dd * Dd);
  cvtw(ffn2_w, Wffn2, (size_t)NBLK * Dd * 4 * Dd);
  // input -> transposed f16
  k_cvtT<<<dim3((Bb * Nn + 255) / 256), dim3(256), 0, stream>>>(x, XtA, Dd);

  const float* hcur = x;
  for (int i = 0; i < NBLK; ++i) {
    const int k = KS[i], dil = DIL[i];

    // ---- fc1: (D x D), BN -> G (f32, feeds normalize + mrconv)
    k_gemm_bn<<<dim3(NT32, Dd / 32, Bb), blk32, 0, stream>>>(
        Wfc1 + (size_t)i * Dd * Dd, XtA, fc1_b + (size_t)i * Dd,
        fc1_bn + (size_t)i * 4 * Dd, nullptr, G, nullptr, Dd, Dd, 0);

    // ---- kNN: normalize -> per-batch Gram dist -> dilated top-k
    k_normalize<<<dim3((Bb * Nn + 127) / 128), dim3(128), 0, stream>>>(G, Gt, sq);
    for (int b = 0; b < Bb; ++b) {
      k_dist<<<dim3(NT32, NT32), blk32, 0, stream>>>(
          Gt + (size_t)b * Nn * Dd, sq + (size_t)b * Nn, dist);
      k_topk<<<dim3((Nn + 127) / 128), dim3(128), 0, stream>>>(
          dist, idxb + (size_t)b * Nn * KMAX, k, dil);
    }

    // ---- MRConv aggregate -> Tt (B, N, 2D) f16
    k_mrconv<<<dim3((Bb * Dd * Nn + 255) / 256), dim3(256), 0, stream>>>(
        G, idxb, Tt, k);

    // ---- gc: (2D x 2D), BN, GELU -> Yt (f16 only)
    k_gemm_bn<<<dim3(NT32, 2 * Dd / 32, Bb), blk32, 0, stream>>>(
        Wgc + (size_t)i * 2 * Dd * 2 * Dd, Tt, gc_b + (size_t)i * 2 * Dd,
        gc_bn + (size_t)i * 4 * 2 * Dd, nullptr, nullptr, Yt, 2 * Dd, 2 * Dd, 1);

    // ---- fc2: (D x 2D), BN, + residual hcur -> hB (f32) + HtB (f16)
    k_gemm_bn<<<dim3(NT32, Dd / 32, Bb), blk32, 0, stream>>>(
        Wfc2 + (size_t)i * Dd * 2 * Dd, Yt, fc2_b + (size_t)i * Dd,
        fc2_bn + (size_t)i * 4 * Dd, hcur, hB, HtB, Dd, 2 * Dd, 0);

    // ---- ffn1: (4D x D), BN, GELU -> Ft (f16 only)
    k_gemm_bn<<<dim3(NT32, 4 * Dd / 32, Bb), blk32, 0, stream>>>(
        Wffn1 + (size_t)i * 4 * Dd * Dd, HtB, ffn1_b + (size_t)i * 4 * Dd,
        ffn1_bn + (size_t)i * 4 * 4 * Dd, nullptr, nullptr, Ft, 4 * Dd, Dd, 1);

    // ---- ffn2: (D x 4D), BN, + residual hB -> hA/d_out (+ XtA for next fc1)
    float*    outp  = (i == NBLK - 1) ? (float*)d_out : hA;
    _Float16* outTp = (i == NBLK - 1) ? nullptr : XtA;
    k_gemm_bn<<<dim3(NT32, Dd / 32, Bb), blk32, 0, stream>>>(
        Wffn2 + (size_t)i * Dd * 4 * Dd, Ft, ffn2_b + (size_t)i * Dd,
        ffn2_bn + (size_t)i * 4 * Dd, hB, outp, outTp, Dd, 4 * Dd, 0);

    hcur = hA;
  }
}
